// MultiHeadCrossAttention_43679817400637
// MI455X (gfx1250) — matrix operand order repaired and checked by reading, not compiled, not run on hardware
//
#include <hip/hip_runtime.h>
#include <hip/hip_bf16.h>
#include <stdint.h>

// Problem constants (match reference)
#define BATCH 2
#define SQ 2048
#define SKV 2048
#define DMODEL 1024
#define NHEAD 16
#define DHEAD 64

typedef __attribute__((ext_vector_type(16))) __bf16 bf16x16;
typedef __attribute__((ext_vector_type(8)))  float  floatx8;
typedef __attribute__((address_space(3))) unsigned short lds_u16;

union BfOp { unsigned int u[8]; bf16x16 v; };

#ifndef __has_builtin
#define __has_builtin(x) 0
#endif
#if __has_builtin(__builtin_amdgcn_permlane16)
#define USE_PERMLANE16 1
#else
#define USE_PERMLANE16 0
#endif

// xor-butterfly within each 16-lane row (lanes 0-15 / 16-31 independently).
// Uses v_permlane16_b32 (pure VALU) when available, else ds_bpermute fallback.
template <int STEP>
static __device__ __forceinline__ float xor16(float x) {
#if USE_PERMLANE16
    constexpr unsigned lo[4] = {0x67452301u, 0x54761032u, 0x32107654u, 0xFEDCBA98u};
    constexpr unsigned hi[4] = {0xEFCDAB89u, 0xDCFE98BAu, 0xBA98FEDCu, 0x76543210u};
    unsigned u = __float_as_uint(x);
    u = __builtin_amdgcn_permlane16(u, u, lo[STEP], hi[STEP], false, false);
    return __uint_as_float(u);
#else
    return __shfl_xor(x, 1 << STEP, 32);
#endif
}

static __device__ __forceinline__ unsigned short f32_to_bf16_bits(float f) {
    unsigned int u = __float_as_uint(f);
    u += 0x7FFFu + ((u >> 16) & 1u);   // round-to-nearest-even
    return (unsigned short)(u >> 16);
}

// Async direct-to-LDS 128-bit copy (CDNA5): per-lane LDS dest (VDST) + global addr.
static __device__ __forceinline__ void async_copy_b128(unsigned lds_off,
                                                       const void* gaddr) {
    asm volatile("global_load_async_to_lds_b128 %0, %1, off"
                 :: "v"(lds_off), "v"(gaddr)
                 : "memory");
}
static __device__ __forceinline__ void wait_asynccnt0() {
    asm volatile("s_wait_asynccnt 0x0" ::: "memory");
}

// ---------------------------------------------------------------------------
// fp32 -> bf16 conversion (plain)
// ---------------------------------------------------------------------------
__global__ void cvt_f32_bf16(const float* __restrict__ src,
                             unsigned short* __restrict__ dst, int n) {
    int i = blockIdx.x * blockDim.x + threadIdx.x;
    if (i < n) dst[i] = f32_to_bf16_bits(src[i]);
}

// ---------------------------------------------------------------------------
// fp32 -> bf16 conversion with transpose: dst[n*K+k] = src[k*N+n]
// ---------------------------------------------------------------------------
__global__ void cvt_f32_bf16_T(const float* __restrict__ src,
                               unsigned short* __restrict__ dst,
                               int K, int N) {
    size_t idx = (size_t)blockIdx.x * blockDim.x + threadIdx.x;
    if (idx >= (size_t)K * N) return;
    int n = (int)(idx / K);
    int k = (int)(idx % K);
    dst[idx] = f32_to_bf16_bits(src[(size_t)k * N + n]);
}

// ---------------------------------------------------------------------------
// V transpose: kv_proj[b*SKV+s][DMODEL + h*64 + d] -> Vt[((b*H+h)*64+d)*SKV+s]
// ---------------------------------------------------------------------------
__global__ void scatter_v(const unsigned short* __restrict__ kvp,
                          unsigned short* __restrict__ Vt) {
    size_t idx = (size_t)blockIdx.x * blockDim.x + threadIdx.x;
    if (idx >= (size_t)BATCH * SKV * DMODEL) return;
    int d = idx & 63;
    int h = (idx >> 6) & 15;
    int s = (int)((idx >> 10) & (SKV - 1));
    int b = (int)(idx >> 21);
    unsigned short val =
        kvp[((size_t)b * SKV + s) * (2 * DMODEL) + DMODEL + h * DHEAD + d];
    Vt[(((size_t)b * NHEAD + h) * DHEAD + d) * SKV + s] = val;
}

// ---------------------------------------------------------------------------
// Tiled bf16 WMMA GEMM: C[M,N] = A[M,K] * Bt[N,K]^T + bias[N]
// Block tile 128x128x64, 256 threads (8 waves), wave tile 32x64.
// Tiles staged via global_load_async_to_lds_b128 (ASYNCcnt path).
// ---------------------------------------------------------------------------
#define BM 128
#define BN 128
#define BKT 64

template <bool BF16_OUT>
__global__ __launch_bounds__(256) void gemm_bf16_wmma(
    const unsigned short* __restrict__ A,   // [M,K] bf16, row-major
    const unsigned short* __restrict__ Bt,  // [N,K] bf16, row-major (pre-transposed)
    const float* __restrict__ bias,         // [N]
    void* __restrict__ Cout,                // bf16 or f32 [M,N]
    int M, int N, int K) {
    __shared__ unsigned short As[BM][BKT];  // 16 KB
    __shared__ unsigned short Bs[BN][BKT];  // 16 KB

    const int tid  = threadIdx.x;
    const int lane = tid & 31;
    const int wid  = tid >> 5;
    const int waveM = wid & 3;   // 4 waves along M  (4*32 = 128)
    const int waveN = wid >> 2;  // 2 waves along N  (2*64 = 128)
    const int hb = lane >> 4;    // lane half
    const int ln = lane & 15;

    const int rowBase = blockIdx.y * BM;
    const int colBase = blockIdx.x * BN;

    floatx8 acc[2][4];
#pragma unroll
    for (int tm = 0; tm < 2; ++tm)
#pragma unroll
        for (int tn = 0; tn < 4; ++tn) acc[tm][tn] = (floatx8)0.0f;

    const uint4* Ag = (const uint4*)A;   // 8 bf16 per uint4
    const uint4* Bg = (const uint4*)Bt;
    const int Kq = K >> 3;               // uint4 per row

    // Per-thread staging coordinates (4 b128 copies per tile per matrix)
    const int sr = tid >> 3;             // row within tile (for i-th copy: +32*i)
    const int sc = tid & 7;              // uint4 column

    for (int k0 = 0; k0 < K; k0 += BKT) {
        // Stage A and B tiles: async direct-to-LDS b128 copies
#pragma unroll
        for (int i = 0; i < 4; ++i) {
            int r = sr + 32 * i;
            unsigned la = (unsigned)(unsigned long long)(lds_u16*)&As[r][8 * sc];
            async_copy_b128(la, &Ag[(size_t)(rowBase + r) * Kq + (k0 >> 3) + sc]);
            unsigned lb = (unsigned)(unsigned long long)(lds_u16*)&Bs[r][8 * sc];
            async_copy_b128(lb, &Bg[(size_t)(colBase + r) * Kq + (k0 >> 3) + sc]);
        }
        // Prefetch next k-tile toward the WGP (global_prefetch_b8)
        if (k0 + BKT < K) {
            __builtin_prefetch(&Ag[(size_t)(rowBase + sr) * Kq + ((k0 + BKT) >> 3) + sc], 0, 1);
            __builtin_prefetch(&Bg[(size_t)(colBase + sr) * Kq + ((k0 + BKT) >> 3) + sc], 0, 1);
        }
        wait_asynccnt0();
        __syncthreads();

#pragma unroll
        for (int kc = 0; kc < 2; ++kc) {   // two K=32 chunks
            BfOp aop[2], bop[4];
#pragma unroll
            for (int tm = 0; tm < 2; ++tm) {
                int m = waveM * 32 + tm * 16 + ln;
                const unsigned int* Ar = (const unsigned int*)&As[m][kc * 32];
#pragma unroll
                for (int v = 0; v < 8; ++v) {
                    int kd = (v < 4) ? (hb * 8 + 2 * v) : (16 + hb * 8 + 2 * (v - 4));
                    aop[tm].u[v] = Ar[kd >> 1];
                }
            }
#pragma unroll
            for (int tn = 0; tn < 4; ++tn) {
                int n = waveN * 64 + tn * 16 + ln;
                const unsigned int* Br = (const unsigned int*)&Bs[n][kc * 32];
#pragma unroll
                for (int v = 0; v < 8; ++v) bop[tn].u[v] = Br[(v < 4) ? (hb * 4 + v) : (8 + hb * 4 + (v - 4))];
            }
#pragma unroll
            for (int tm = 0; tm < 2; ++tm)
#pragma unroll
                for (int tn = 0; tn < 4; ++tn)
                    acc[tm][tn] = __builtin_amdgcn_wmma_f32_16x16x32_bf16(
                        false, aop[tm].v, false, bop[tn].v, (short)0, acc[tm][tn],
                        false, false);
        }
        __syncthreads();
    }

    // Epilogue: bias + store
#pragma unroll
    for (int tm = 0; tm < 2; ++tm) {
#pragma unroll
        for (int tn = 0; tn < 4; ++tn) {
            int col = colBase + waveN * 64 + tn * 16 + ln;
            float bv = bias[col];
#pragma unroll
            for (int i = 0; i < 8; ++i) {
                int row = rowBase + waveM * 32 + tm * 16 + i + 8 * hb;
                float val = acc[tm][tn][i] + bv;
                if (BF16_OUT)
                    ((unsigned short*)Cout)[(size_t)row * N + col] =
                        f32_to_bf16_bits(val);
                else
                    ((float*)Cout)[(size_t)row * N + col] = val;
            }
        }
    }
}

// ---------------------------------------------------------------------------
// Flash attention: 1 block per (b, h, 64 q-rows); 4 waves x 16 rows each.
// Per 64-wide KV tile: 8 score WMMAs, online softmax (VALU permlane16
// butterflies), P restaged via LDS, 8 PV WMMAs.
// ---------------------------------------------------------------------------
__global__ __launch_bounds__(128) void attn_flash(
    const unsigned short* __restrict__ Qp,   // [B*SQ, D] bf16 (cols h*64+d)
    const unsigned short* __restrict__ KVp,  // [B*SKV, 2D] bf16 (K in cols h*64+d)
    const unsigned short* __restrict__ Vt,   // [B,H,DH,SKV] bf16
    unsigned short* __restrict__ Out) {      // [B*SQ, D] bf16
    const int QT = SQ / 64;
    int qt = blockIdx.x % QT;
    int h  = (blockIdx.x / QT) % NHEAD;
    int b  = blockIdx.x / (QT * NHEAD);

    const int lane = threadIdx.x & 31;
    const int wave = threadIdx.x >> 5;
    const int hb = lane >> 4;
    const int ln = lane & 15;
    const int q0 = qt * 64 + wave * 16;

    __shared__ unsigned short Pl[4][16][64];   // 8 KB

    // Load resident Q A-operands (d 0..31 and d 32..63)
    BfOp aq[2];
    {
        int m = q0 + ln;
        const unsigned int* Qr =
            (const unsigned int*)(Qp + ((size_t)(b * SQ + m)) * DMODEL + h * DHEAD);
#pragma unroll
        for (int half = 0; half < 2; ++half)
#pragma unroll
            for (int v = 0; v < 8; ++v) {
                int kd = half * 32 +
                         ((v < 4) ? (hb * 8 + 2 * v) : (16 + hb * 8 + 2 * (v - 4)));
                aq[half].u[v] = Qr[kd >> 1];
            }
    }

    float mrun[8], lrun[8];
    floatx8 acc[4];
#pragma unroll
    for (int i = 0; i < 8; ++i) { mrun[i] = -1e30f; lrun[i] = 0.0f; }
#pragma unroll
    for (int t = 0; t < 4; ++t) acc[t] = (floatx8)0.0f;

    const float scale = 0.125f;  // 1/sqrt(64)

    for (int j = 0; j < SKV; j += 64) {
        // ---- scores S[16q x 64kv] as four 16x16 tiles ----
        floatx8 s[4];
#pragma unroll
        for (int tn = 0; tn < 4; ++tn) s[tn] = (floatx8)0.0f;
#pragma unroll
        for (int tn = 0; tn < 4; ++tn) {
            int kv = j + tn * 16 + ln;
            const unsigned int* Kr =
                (const unsigned int*)(KVp + ((size_t)(b * SKV + kv)) * (2 * DMODEL) +
                                      h * DHEAD);
#pragma unroll
            for (int half = 0; half < 2; ++half) {
                BfOp bk;
#pragma unroll
                for (int v = 0; v < 8; ++v) {
                    int kd = half * 32 +
                             ((v < 4) ? (hb * 8 + 2 * v) : (16 + hb * 8 + 2 * (v - 4)));
                    bk.u[v] = Kr[kd >> 1];
                }
                s[tn] = __builtin_amdgcn_wmma_f32_16x16x32_bf16(
                    false, aq[half].v, false, bk.v, (short)0, s[tn], false, false);
            }
        }

        // ---- online softmax update (fold 4 local values, then butterfly) ----
#pragma unroll
        for (int i = 0; i < 8; ++i) {
            float v0 = s[0][i] * scale;
            float v1 = s[1][i] * scale;
            float v2 = s[2][i] * scale;
            float v3 = s[3][i] * scale;
            float mx = fmaxf(fmaxf(v0, v1), fmaxf(v2, v3));
            mx = fmaxf(mx, xor16<0>(mx));
            mx = fmaxf(mx, xor16<1>(mx));
            mx = fmaxf(mx, xor16<2>(mx));
            mx = fmaxf(mx, xor16<3>(mx));
            float mnew = fmaxf(mrun[i], mx);
            float p0 = __expf(v0 - mnew);
            float p1 = __expf(v1 - mnew);
            float p2 = __expf(v2 - mnew);
            float p3 = __expf(v3 - mnew);
            float corr = __expf(mrun[i] - mnew);
            float rs = (p0 + p1) + (p2 + p3);
            rs += xor16<0>(rs);
            rs += xor16<1>(rs);
            rs += xor16<2>(rs);
            rs += xor16<3>(rs);
            lrun[i] = lrun[i] * corr + rs;
            mrun[i] = mnew;
#pragma unroll
            for (int t = 0; t < 4; ++t) acc[t][i] *= corr;
            int r = i + 8 * hb;
            Pl[wave][r][ln]      = f32_to_bf16_bits(p0);
            Pl[wave][r][16 + ln] = f32_to_bf16_bits(p1);
            Pl[wave][r][32 + ln] = f32_to_bf16_bits(p2);
            Pl[wave][r][48 + ln] = f32_to_bf16_bits(p3);
        }
        __syncthreads();

        // ---- rebuild P as two A-operands (16x64) ----
        BfOp ap[2];
        {
            const unsigned int* Pr = (const unsigned int*)&Pl[wave][ln][0];
#pragma unroll
            for (int half = 0; half < 2; ++half)
#pragma unroll
                for (int v = 0; v < 8; ++v) {
                    int kd = half * 32 +
                             ((v < 4) ? (hb * 8 + 2 * v) : (16 + hb * 8 + 2 * (v - 4)));
                    ap[half].u[v] = Pr[kd >> 1];
                }
        }
        __syncthreads();

        // ---- O += P * V : 8 WMMAs over 4 d-tiles x 2 kv-halves ----
#pragma unroll
        for (int t = 0; t < 4; ++t) {
            int d = t * 16 + ln;
            const unsigned int* Vr =
                (const unsigned int*)(Vt +
                                      ((((size_t)b * NHEAD + h) * DHEAD + d)) * SKV + j);
#pragma unroll
            for (int half = 0; half < 2; ++half) {
                BfOp bv;
#pragma unroll
                for (int v = 0; v < 8; ++v) bv.u[v] = Vr[half * 16 + ((v < 4) ? (hb * 4 + v) : (8 + hb * 4 + (v - 4)))];
                acc[t] = __builtin_amdgcn_wmma_f32_16x16x32_bf16(
                    false, ap[half].v, false, bv.v, (short)0, acc[t], false, false);
            }
        }
    }

    // ---- finalize: normalize (one reciprocal per row) and store bf16 ----
#pragma unroll
    for (int i = 0; i < 8; ++i) {
        float inv = 1.0f / lrun[i];
        int row = q0 + i + 8 * hb;
#pragma unroll
        for (int t = 0; t < 4; ++t) {
            int col = h * DHEAD + t * 16 + ln;
            Out[((size_t)(b * SQ + row)) * DMODEL + col] =
                f32_to_bf16_bits(acc[t][i] * inv);
        }
    }
}

// ---------------------------------------------------------------------------
// Host-side launcher
// ---------------------------------------------------------------------------
extern "C" void kernel_launch(void* const* d_in, const int* in_sizes, int n_in,
                              void* d_out, int out_size, void* d_ws, size_t ws_size,
                              hipStream_t stream) {
    const float* query = (const float*)d_in[0];  // [B,SQ,D]
    const float* kvin  = (const float*)d_in[1];  // [B,SKV,D]
    const float* Wq    = (const float*)d_in[2];  // [D,D]
    const float* bq    = (const float*)d_in[3];  // [D]
    const float* Wkv   = (const float*)d_in[4];  // [D,2D]
    const float* bkv   = (const float*)d_in[5];  // [2D]
    const float* Wo    = (const float*)d_in[6];  // [D,D]
    const float* bo    = (const float*)d_in[7];  // [D]
    float* out = (float*)d_out;

    // Workspace layout (ushort elements)
    unsigned short* ws = (unsigned short*)d_ws;
    const size_t nXq   = (size_t)BATCH * SQ * DMODEL;       // 4,194,304
    const size_t nXkv  = (size_t)BATCH * SKV * DMODEL;      // 4,194,304
    const size_t nWq   = (size_t)DMODEL * DMODEL;           // 1,048,576
    const size_t nWkv  = (size_t)DMODEL * 2 * DMODEL;       // 2,097,152
    const size_t nWo   = nWq;
    const size_t nQp   = nXq;
    const size_t nKVp  = (size_t)BATCH * SKV * 2 * DMODEL;  // 8,388,608
    const size_t nVt   = nXkv;

    unsigned short* Xq_b   = ws;                // bf16 query input
    unsigned short* Xkv_b  = Xq_b + nXq;        // bf16 kv input
    unsigned short* WqT    = Xkv_b + nXkv;      // [N,K] bf16
    unsigned short* WkvT   = WqT + nWq;         // [2D, D] bf16
    unsigned short* WoT    = WkvT + nWkv;       // [N,K] bf16
    unsigned short* Qproj  = WoT + nWo;         // [B*SQ, D] bf16
    unsigned short* KVproj = Qproj + nQp;       // [B*SKV, 2D] bf16
    unsigned short* Vt     = KVproj + nKVp;     // [B,H,DH,SKV] bf16
    unsigned short* AttO   = Vt + nVt;          // [B*SQ, D] bf16
    (void)ws_size; (void)n_in; (void)in_sizes; (void)out_size;

    const int T = 256;
    // 1) fp32 -> bf16 staging (weights transposed to [N][K])
    cvt_f32_bf16<<<(int)((nXq  + T - 1) / T), T, 0, stream>>>(query, Xq_b,  (int)nXq);
    cvt_f32_bf16<<<(int)((nXkv + T - 1) / T), T, 0, stream>>>(kvin,  Xkv_b, (int)nXkv);
    cvt_f32_bf16_T<<<(int)((nWq  + T - 1) / T), T, 0, stream>>>(Wq,  WqT,  DMODEL, DMODEL);
    cvt_f32_bf16_T<<<(int)((nWkv + T - 1) / T), T, 0, stream>>>(Wkv, WkvT, DMODEL, 2 * DMODEL);
    cvt_f32_bf16_T<<<(int)((nWo  + T - 1) / T), T, 0, stream>>>(Wo,  WoT,  DMODEL, DMODEL);

    // 2) Q projection: [4096,1024] x [1024,1024]
    {
        dim3 grid(DMODEL / BN, (BATCH * SQ) / BM);
        gemm_bf16_wmma<true><<<grid, 256, 0, stream>>>(
            Xq_b, WqT, bq, Qproj, BATCH * SQ, DMODEL, DMODEL);
    }
    // 3) KV projection: [4096,1024] x [1024,2048]
    {
        dim3 grid((2 * DMODEL) / BN, (BATCH * SKV) / BM);
        gemm_bf16_wmma<true><<<grid, 256, 0, stream>>>(
            Xkv_b, WkvT, bkv, KVproj, BATCH * SKV, 2 * DMODEL, DMODEL);
    }
    // 4) V transpose scatter
    scatter_v<<<(int)((nXkv + T - 1) / T), T, 0, stream>>>(KVproj, Vt);

    // 5) Flash attention: B*H*(SQ/64) blocks x 128 threads
    attn_flash<<<BATCH * NHEAD * (SQ / 64), 128, 0, stream>>>(Qproj, KVproj, Vt, AttO);

    // 6) Output projection -> fp32 d_out
    {
        dim3 grid(DMODEL / BN, (BATCH * SQ) / BM);
        gemm_bf16_wmma<false><<<grid, 256, 0, stream>>>(
            AttO, WoT, bo, out, BATCH * SQ, DMODEL, DMODEL);
    }
}
